// AeroNet_42683384987909
// MI455X (gfx1250) — compile-verified, matrix operand order checked by reading, not run on hardware
//
#include <hip/hip_runtime.h>
#include <math.h>

// ============================================================================
// AeroNet (PointNet++ + transformer bottleneck) for MI455X / gfx1250.
// All dense math goes through one WMMA GEMM kernel:
//   C = act(scale * (A @ B^T) + bias) [+C]
// using v_wmma_f32_16x16x32_f16 (f16 in, f32 accumulate).
// Each wave owns a 32x32 output tile -> 4 independent WMMA accumulator chains
// per k-step (hides the F16 WMMA RAW latency, 2x flops per byte loaded).
// All GEMM operands are padded in the workspace (K to 32, rows to 32) so the
// fragment loads are unconditional global_load_b128 — no divergent bounds
// checks in the hot loop. Misaligned weights are pre-packed once.
// Scalar/latency stages (FPS, ball query, 3-NN, LN, softmax, GN, maxpool)
// are plain VALU kernels.
// ============================================================================

typedef __attribute__((ext_vector_type(16))) _Float16 v16h;
typedef __attribute__((ext_vector_type(8)))  float    v8f;

#define DEV __device__ __forceinline__

DEV float an_gelu(float x) { return 0.5f * x * (1.0f + erff(x * 0.70710678118654752f)); }

// ---------------------------------------------------------------------------
// WMMA fragment loader. 16-bit A-matrix 16x32 layout (CDNA5 ISA):
//   lane in [0,15]  -> row M=lane,    K = k0+{0..7, 16..23}
//   lane in [16,31] -> row M=lane-16, K = k0+8+{0..7, 16..23}
// B (32x16 KxN) is the transpose-symmetric mapping, so loading B^T (N x K)
// with the same pattern is a valid B operand. `p` must point at
// row(r0+lane15)*ld + k0 + lanehalf*8, 16B aligned, fully in-bounds (padded).
// ---------------------------------------------------------------------------
DEV v16h an_cvt_frag(const float* __restrict__ p) {
  const float4* q = (const float4*)p;
  float4 a0 = q[0];          // K +0..3
  float4 a1 = q[1];          // K +4..7
  float4 b0 = q[4];          // K +16..19
  float4 b1 = q[5];          // K +20..23
  v16h f;
  f[0]  = (_Float16)a0.x; f[1]  = (_Float16)a0.y; f[2]  = (_Float16)a0.z; f[3]  = (_Float16)a0.w;
  f[4]  = (_Float16)a1.x; f[5]  = (_Float16)a1.y; f[6]  = (_Float16)a1.z; f[7]  = (_Float16)a1.w;
  f[8]  = (_Float16)b0.x; f[9]  = (_Float16)b0.y; f[10] = (_Float16)b0.z; f[11] = (_Float16)b0.w;
  f[12] = (_Float16)b1.x; f[13] = (_Float16)b1.y; f[14] = (_Float16)b1.z; f[15] = (_Float16)b1.w;
  return f;
}

// flags: bit1 = accumulate into C, bit2 = gelu activation
// Block tile 64(M) x 128(N): 8 waves as 2x4, each wave a 32x32 tile.
__global__ void an_gemm(const float* __restrict__ A, int lda,
                        const float* __restrict__ B, int ldb,
                        float* __restrict__ C, int ldc,
                        int M, int N, int K,
                        const float* __restrict__ bias, float scale, int flags) {
  int lane = threadIdx.x & 31;
  int wave = threadIdx.x >> 5;
  int mt = blockIdx.y * 64 + (wave >> 2) * 32;
  int nt = blockIdx.x * 128 + (wave & 3) * 32;
  if (mt >= M || nt >= N) return;  // wave-uniform: EXEC stays all-ones for WMMA
  int l15 = lane & 15, kh = (lane >> 4) << 3;
  const float* ap0 = A + (size_t)(mt + l15) * lda + kh;
  const float* ap1 = ap0 + (size_t)16 * lda;
  const float* bp0 = B + (size_t)(nt + l15) * ldb + kh;
  const float* bp1 = bp0 + (size_t)16 * ldb;
  v8f acc00 = {}, acc01 = {}, acc10 = {}, acc11 = {};
  int kiters = (K + 31) >> 5;
  for (int it = 0; it < kiters; ++it) {
    __builtin_prefetch(ap0 + 64, 0, 1);   // gfx1250 global_prefetch_b8
    __builtin_prefetch(bp0 + 64, 0, 1);
    v16h a0 = an_cvt_frag(ap0);
    v16h a1 = an_cvt_frag(ap1);
    v16h b0 = an_cvt_frag(bp0);
    v16h b1 = an_cvt_frag(bp1);
    acc00 = __builtin_amdgcn_wmma_f32_16x16x32_f16(false, a0, false, b0, (short)0, acc00, false, false);
    acc01 = __builtin_amdgcn_wmma_f32_16x16x32_f16(false, a0, false, b1, (short)0, acc01, false, false);
    acc10 = __builtin_amdgcn_wmma_f32_16x16x32_f16(false, a1, false, b0, (short)0, acc10, false, false);
    acc11 = __builtin_amdgcn_wmma_f32_16x16x32_f16(false, a1, false, b1, (short)0, acc11, false, false);
    ap0 += 32; ap1 += 32; bp0 += 32; bp1 += 32;
  }
  // C/D layout per 16x16 tile: lane<16 -> M=+r, N=+lane ; lane>=16 -> M=+8+r
#pragma unroll
  for (int ti = 0; ti < 2; ++ti) {
#pragma unroll
    for (int tj = 0; tj < 2; ++tj) {
      v8f acc = (ti == 0) ? (tj == 0 ? acc00 : acc01) : (tj == 0 ? acc10 : acc11);
      int n    = nt + tj * 16 + l15;
      int mrow = mt + ti * 16 + kh;
      float bv = (bias && n < N) ? bias[n] : 0.0f;
#pragma unroll
      for (int r = 0; r < 8; ++r) {
        int m = mrow + r;
        if (m < M && n < N) {
          float v = acc[r] * scale + bv;
          if (flags & 4) v = an_gelu(v);
          size_t o = (size_t)m * ldc + n;
          if (flags & 2) C[o] += v; else C[o] = v;
        }
      }
    }
  }
}

// Pack weight [N x K] row-major into padded [align32(N) x Kp], zero filled.
__global__ void an_packw(const float* __restrict__ W, int N, int K,
                         float* __restrict__ out, int Kp, long total) {
  for (long i = blockIdx.x * (long)blockDim.x + threadIdx.x; i < total;
       i += (long)gridDim.x * blockDim.x) {
    long r = i / Kp; int c = (int)(i % Kp);
    out[i] = (r < N && c < K) ? W[(size_t)r * K + c] : 0.0f;
  }
}

// Transpose V heads out of qkv: vt[(h*64+d)*Kp + t] = qkv[t*ldq + 1024 + h*64 + d]
__global__ void an_vt(const float* __restrict__ qkv, int ldq, int T, int Kp,
                      float* __restrict__ vt) {
  long total = 512L * Kp;
  for (long i = blockIdx.x * (long)blockDim.x + threadIdx.x; i < total;
       i += (long)gridDim.x * blockDim.x) {
    int row = (int)(i / Kp), t = (int)(i % Kp);
    vt[i] = (t < T) ? qkv[(size_t)t * ldq + 1024 + row] : 0.0f;
  }
}

// ---------------------------------------------------------------------------
// Farthest point sampling: one persistent block, LDS argmax reduction.
// ---------------------------------------------------------------------------
__global__ void an_fps(const float* __restrict__ xyz, int N, int M,
                       int* __restrict__ out, float* __restrict__ dist) {
  __shared__ float sv[256];
  __shared__ int   si[256];
  __shared__ int   sfar;
  int tid = threadIdx.x;
  for (int i = tid; i < N; i += 256) dist[i] = 1e10f;
  if (tid == 0) sfar = 0;
  __syncthreads();
  for (int it = 0; it < M; ++it) {
    int far = sfar;
    if (tid == 0) out[it] = far;
    float cx = xyz[far * 3], cy = xyz[far * 3 + 1], cz = xyz[far * 3 + 2];
    float best = -1.0f; int bi = 0;
    for (int i = tid; i < N; i += 256) {
      float dx = xyz[i * 3] - cx, dy = xyz[i * 3 + 1] - cy, dz = xyz[i * 3 + 2] - cz;
      float dd = fminf(dist[i], dx * dx + dy * dy + dz * dz);
      dist[i] = dd;
      if (dd > best) { best = dd; bi = i; }
    }
    sv[tid] = best; si[tid] = bi;
    __syncthreads();
    for (int s = 128; s > 0; s >>= 1) {
      if (tid < s && sv[tid + s] > sv[tid]) { sv[tid] = sv[tid + s]; si[tid] = si[tid + s]; }
      __syncthreads();
    }
    if (tid == 0) sfar = si[0];
    __syncthreads();
  }
}

__global__ void an_gather_xyz(const float* __restrict__ xyz, const int* __restrict__ idx,
                              float* __restrict__ out, int n) {
  int i = blockIdx.x * blockDim.x + threadIdx.x;
  if (i < n) {
    int j = idx[i];
    out[i * 3] = xyz[j * 3]; out[i * 3 + 1] = xyz[j * 3 + 1]; out[i * 3 + 2] = xyz[j * 3 + 2];
  }
}

// Ball query + grouping: first k in-range indices (ascending), pad with first.
// Rows [cent*k + j] of width ldg (zero pre-filled): relative xyz ++ features.
__global__ void an_ball_group(const float* __restrict__ xyz, const float* __restrict__ feat,
                              int Cf, const float* __restrict__ nxyz, int ncent, int N,
                              float r2, int k, float* __restrict__ out, int ldg) {
  int c = blockIdx.x * blockDim.x + threadIdx.x;
  if (c >= ncent) return;
  float cx = nxyz[c * 3], cy = nxyz[c * 3 + 1], cz = nxyz[c * 3 + 2];
  int cnt = 0, first = N - 1;
  for (int j = 0; j < N && cnt < k; ++j) {
    float dx = xyz[j * 3] - cx, dy = xyz[j * 3 + 1] - cy, dz = xyz[j * 3 + 2] - cz;
    if (dx * dx + dy * dy + dz * dz <= r2) {
      if (cnt == 0) first = j;
      float* row = out + (size_t)(c * k + cnt) * ldg;
      row[0] = dx; row[1] = dy; row[2] = dz;
      for (int ch = 0; ch < Cf; ++ch) row[3 + ch] = feat[(size_t)j * Cf + ch];
      ++cnt;
    }
  }
  for (int q = cnt; q < k; ++q) {
    int j = first;
    float* row = out + (size_t)(c * k + q) * ldg;
    row[0] = xyz[j * 3] - cx; row[1] = xyz[j * 3 + 1] - cy; row[2] = xyz[j * 3 + 2] - cz;
    for (int ch = 0; ch < Cf; ++ch) row[3 + ch] = feat[(size_t)j * Cf + ch];
  }
}

__global__ void an_maxpool(const float* __restrict__ in, int ldi, int C, int ncent, int k,
                           float* __restrict__ out, int ldo, int coff) {
  long i = blockIdx.x * (long)blockDim.x + threadIdx.x;
  if (i >= (long)ncent * C) return;
  int c = (int)(i / C), ch = (int)(i % C);
  float m = -1e30f;
  for (int j = 0; j < k; ++j) m = fmaxf(m, in[(size_t)(c * k + j) * ldi + ch]);
  out[(size_t)c * ldo + coff + ch] = m;
}

// 3-NN search (per fine point) + weighted interpolation.
__global__ void an_3nn(const float* __restrict__ fx, int NF,
                       const float* __restrict__ cx, int NC,
                       int* __restrict__ id3, float* __restrict__ w3) {
  int p = blockIdx.x * blockDim.x + threadIdx.x;
  if (p >= NF) return;
  float px = fx[p * 3], py = fx[p * 3 + 1], pz = fx[p * 3 + 2];
  float d0 = 3e38f, d1 = 3e38f, d2 = 3e38f; int i0 = 0, i1 = 0, i2 = 0;
  for (int j = 0; j < NC; ++j) {
    float dx = px - cx[j * 3], dy = py - cx[j * 3 + 1], dz = pz - cx[j * 3 + 2];
    float d = dx * dx + dy * dy + dz * dz;
    if (d < d0)      { d2 = d1; i2 = i1; d1 = d0; i1 = i0; d0 = d; i0 = j; }
    else if (d < d1) { d2 = d1; i2 = i1; d1 = d;  i1 = j; }
    else if (d < d2) { d2 = d;  i2 = j; }
  }
  float w0 = 1.0f / (d0 + 1e-8f), w1 = 1.0f / (d1 + 1e-8f), w2 = 1.0f / (d2 + 1e-8f);
  float inv = 1.0f / (w0 + w1 + w2);
  id3[p * 3] = i0; id3[p * 3 + 1] = i1; id3[p * 3 + 2] = i2;
  w3[p * 3] = w0 * inv; w3[p * 3 + 1] = w1 * inv; w3[p * 3 + 2] = w2 * inv;
}

__global__ void an_interp(const float* __restrict__ fc, int C,
                          const int* __restrict__ id3, const float* __restrict__ w3,
                          float* __restrict__ out, long total) {
  for (long i = blockIdx.x * (long)blockDim.x + threadIdx.x; i < total;
       i += (long)gridDim.x * blockDim.x) {
    long p = i / C; int ch = (int)(i % C);
    float s = 0.f;
    for (int j = 0; j < 3; ++j)
      s += w3[p * 3 + j] * fc[(size_t)id3[p * 3 + j] * C + ch];
    out[i] = s;
  }
}

__global__ void an_concat2(const float* __restrict__ A, int C1,
                           const float* __restrict__ B, int C2,
                           float* __restrict__ out, int ldo, long nrows) {
  int Ct = C1 + C2;
  long total = nrows * Ct;
  for (long i = blockIdx.x * (long)blockDim.x + threadIdx.x; i < total;
       i += (long)gridDim.x * blockDim.x) {
    long p = i / Ct; int c = (int)(i % Ct);
    out[p * ldo + c] = (c < C1) ? A[(size_t)p * C1 + c] : B[(size_t)p * C2 + (c - C1)];
  }
}

__global__ void an_copy(const float* __restrict__ a, float* __restrict__ b, long n) {
  for (long i = blockIdx.x * (long)blockDim.x + threadIdx.x; i < n;
       i += (long)gridDim.x * blockDim.x) b[i] = a[i];
}

__global__ void an_gvec(const float* __restrict__ gc, const float* __restrict__ Wg,
                        const float* __restrict__ bg, float* __restrict__ out, int C) {
  int c = blockIdx.x * blockDim.x + threadIdx.x;
  if (c < C) {
    float s = bg[c];
    for (int j = 0; j < 4; ++j) s += gc[j] * Wg[c * 4 + j];
    out[c] = s;
  }
}

__global__ void an_layernorm(const float* __restrict__ x, float* __restrict__ y, int C,
                             const float* __restrict__ g, const float* __restrict__ b) {
  int r = blockIdx.x;
  __shared__ float ss[256], sq[256];
  float s = 0.f, q = 0.f;
  for (int c = threadIdx.x; c < C; c += 256) {
    float v = x[(size_t)r * C + c]; s += v; q += v * v;
  }
  ss[threadIdx.x] = s; sq[threadIdx.x] = q;
  __syncthreads();
  for (int st = 128; st > 0; st >>= 1) {
    if (threadIdx.x < st) { ss[threadIdx.x] += ss[threadIdx.x + st]; sq[threadIdx.x] += sq[threadIdx.x + st]; }
    __syncthreads();
  }
  float m = ss[0] / C, var = sq[0] / C - (ss[0] / C) * (ss[0] / C);
  float inv = rsqrtf(var + 1e-5f);
  for (int c = threadIdx.x; c < C; c += 256)
    y[(size_t)r * C + c] = (x[(size_t)r * C + c] - m) * inv * g[c] + b[c];
}

// rows laid out as [head][rowsPerHead][ld]
__global__ void an_softmax(float* __restrict__ s, int C, int ld,
                           long headStride, int rowsPerHead) {
  int r = blockIdx.x;
  float* row = s + (long)(r / rowsPerHead) * headStride + (long)(r % rowsPerHead) * ld;
  __shared__ float sh[256];
  float m = -1e30f;
  for (int c = threadIdx.x; c < C; c += 256) m = fmaxf(m, row[c]);
  sh[threadIdx.x] = m; __syncthreads();
  for (int st = 128; st > 0; st >>= 1) {
    if (threadIdx.x < st) sh[threadIdx.x] = fmaxf(sh[threadIdx.x], sh[threadIdx.x + st]);
    __syncthreads();
  }
  m = sh[0]; __syncthreads();
  float sum = 0.f;
  for (int c = threadIdx.x; c < C; c += 256) { float e = __expf(row[c] - m); row[c] = e; sum += e; }
  sh[threadIdx.x] = sum; __syncthreads();
  for (int st = 128; st > 0; st >>= 1) {
    if (threadIdx.x < st) sh[threadIdx.x] += sh[threadIdx.x + st];
    __syncthreads();
  }
  float inv = 1.0f / sh[0];
  for (int c = threadIdx.x; c < C; c += 256) row[c] *= inv;
}

__global__ void an_gn_stats(const float* __restrict__ x, int Np, int C, int groups,
                            float* __restrict__ stats) {
  int g = blockIdx.x;
  int cg = C / groups;
  __shared__ float ss[256], sq[256];
  float s = 0.f, q = 0.f;
  long tot = (long)Np * cg;
  for (long i = threadIdx.x; i < tot; i += 256) {
    long r = i / cg; int ch = g * cg + (int)(i % cg);
    float v = x[(size_t)r * C + ch]; s += v; q += v * v;
  }
  ss[threadIdx.x] = s; sq[threadIdx.x] = q;
  __syncthreads();
  for (int st = 128; st > 0; st >>= 1) {
    if (threadIdx.x < st) { ss[threadIdx.x] += ss[threadIdx.x + st]; sq[threadIdx.x] += sq[threadIdx.x + st]; }
    __syncthreads();
  }
  if (threadIdx.x == 0) {
    float m = ss[0] / (float)tot;
    float var = sq[0] / (float)tot - m * m;
    stats[2 * g] = m; stats[2 * g + 1] = rsqrtf(var + 1e-5f);
  }
}

__global__ void an_gn_apply(const float* __restrict__ x, float* __restrict__ y,
                            int Np, int C, int groups, const float* __restrict__ stats,
                            const float* __restrict__ gamma, const float* __restrict__ beta) {
  int cg = C / groups;
  long total = (long)Np * C;
  for (long i = blockIdx.x * (long)blockDim.x + threadIdx.x; i < total;
       i += (long)gridDim.x * blockDim.x) {
    int ch = (int)(i % C);
    int g = ch / cg;
    float v = (x[i] - stats[2 * g]) * stats[2 * g + 1] * gamma[ch] + beta[ch];
    y[i] = an_gelu(v);
  }
}

// ---------------------------------------------------------------------------
// Host side
// ---------------------------------------------------------------------------
static inline int an_al32(int x) { return (x + 31) & ~31; }
static inline int an_nblk(long n) {
  long b = (n + 255) / 256;
  return (int)(b > 65535 ? 65535 : (b < 1 ? 1 : b));
}

static void an_launch_gemm(hipStream_t st, const float* A, int lda, const float* B, int ldb,
                           float* C, int ldc, int M, int N, int K,
                           const float* bias, float scale, int flags) {
  dim3 grid((N + 127) / 128, (M + 63) / 64);
  an_gemm<<<grid, 256, 0, st>>>(A, lda, B, ldb, C, ldc, M, N, K, bias, scale, flags);
}

struct WP { const float* p; int ld; };

extern "C" void kernel_launch(void* const* d_in, const int* in_sizes, int n_in,
                              void* d_out, int out_size, void* d_ws, size_t ws_size,
                              hipStream_t stream) {
  (void)in_sizes; (void)n_in; (void)out_size; (void)ws_size;
  auto F = [&](int i) { return (const float*)d_in[i]; };

  const float* xyz   = F(0);   // [16384,3]
  const float* feats = F(1);   // [16384,3]
  const float* gcond = F(2);   // [4]

  // Params in setup_inputs()/make_params() insertion order, leaves depth-first.
  int pi = 3;
  const float* Wsa1r[2][3]; for (int b = 0; b < 2; ++b) for (int l = 0; l < 3; ++l) Wsa1r[b][l] = F(pi++);
  const float* Wsa2r[2][3]; for (int b = 0; b < 2; ++b) for (int l = 0; l < 3; ++l) Wsa2r[b][l] = F(pi++);
  const float* Wsa3r[2][3]; for (int b = 0; b < 2; ++b) for (int l = 0; l < 3; ++l) Wsa3r[b][l] = F(pi++);
  const float *bin_w = F(pi++), *bin_b = F(pi++);
  const float *bout_w = F(pi++), *bout_b = F(pi++);
  struct BlkP { const float *g1,*b1,*Wqkv,*bqkv,*Wo,*bo,*g2,*b2,*W1,*bf1,*W2,*bf2,*Wg,*bg; } blk[4];
  for (int b = 0; b < 4; ++b) {
    blk[b].g1 = F(pi++); blk[b].b1 = F(pi++);
    blk[b].Wqkv = F(pi++); blk[b].bqkv = F(pi++);
    blk[b].Wo = F(pi++); blk[b].bo = F(pi++);
    blk[b].g2 = F(pi++); blk[b].b2 = F(pi++);
    blk[b].W1 = F(pi++); blk[b].bf1 = F(pi++);
    blk[b].W2 = F(pi++); blk[b].bf2 = F(pi++);
    blk[b].Wg = F(pi++); blk[b].bg = F(pi++);
  }
  const float* fp3r[2] = { F(pi++), F(pi++) };
  const float* fp2r[2] = { F(pi++), F(pi++) };
  const float* fp1r[3] = { F(pi++), F(pi++), F(pi++) };
  const float *h0w = F(pi++), *h0b = F(pi++), *h0g = F(pi++), *h0gb = F(pi++);
  const float *h1w = F(pi++), *h1b = F(pi++), *h1g = F(pi++), *h1gb = F(pi++);
  const float *how = F(pi++), *hob = F(pi++);

  const float BN_S = 0.999995000037f;  // 1/sqrt(1+1e-5)

  // Workspace bump allocator: persistent region first, transient arena after.
  char* wsb = (char*)d_ws;
  size_t off = 0;
  size_t tmp_base = 0;
  auto alloc  = [&](size_t n) { float* p = (float*)(wsb + off); off += ((n * 4 + 255) / 256) * 256; return p; };
  auto alloci = [&](size_t n) { int*   p = (int*)  (wsb + off); off += ((n * 4 + 255) / 256) * 256; return p; };
  auto tmp_reset = [&]() { off = tmp_base; };
  auto zero = [&](void* p, size_t nfloats) { hipMemsetAsync(p, 0, nfloats * 4, stream); };

  // Weight packing: only needed when N%32 or K%32 != 0; else use in place.
  auto packw = [&](const float* W, int N, int K) -> WP {
    if ((N & 31) == 0 && (K & 31) == 0) return { W, K };
    int Kp = an_al32(K), Np = an_al32(N);
    float* o = alloc((size_t)Np * Kp);
    long tot = (long)Np * Kp;
    an_packw<<<an_nblk(tot), 256, 0, stream>>>(W, N, K, o, Kp, tot);
    return { o, Kp };
  };

  // Persistent buffers
  float* dist = alloc(16384);
  int* idx1 = alloci(4096); int* idx2 = alloci(1024); int* idx3p = alloci(256);
  float* l1x = alloc(4096 * 3); float* l2x = alloc(1024 * 3); float* l3x = alloc(256 * 3);
  float* l1f = alloc((size_t)4096 * 128);
  float* l2f = alloc((size_t)1024 * 256);
  float* l3f = alloc((size_t)256 * 512);
  float* tok = alloc((size_t)256 * 512);
  float* xg0 = alloc((size_t)288 * 512);
  float* xg1 = alloc((size_t)288 * 512);
  float* l3fn = alloc((size_t)256 * 512);
  float* l2fn = alloc((size_t)1024 * 256);
  float* l1fn = alloc((size_t)4096 * 128);
  float* l0f  = alloc((size_t)16384 * 128);
  float* t1   = alloc((size_t)16384 * 128);
  float* t2   = alloc((size_t)16384 * 64);
  float* gnst = alloc(16);

  // Packed weights (persistent)
  WP Wsa1[2][3], Wsa2[2][3], Wsa3[2][3];
  for (int b = 0; b < 2; ++b) for (int l = 0; l < 3; ++l) {
    const int co1[3] = {32, 32, 64}, ci1[3] = {6, 32, 32};
    Wsa1[b][l] = packw(Wsa1r[b][l], co1[l], ci1[l]);
  }
  {
    const int co2a[3] = {64, 64, 128}, ci2a[3] = {131, 64, 64};
    const int co2b[3] = {64, 96, 128}, ci2b[3] = {131, 64, 96};
    for (int l = 0; l < 3; ++l) { Wsa2[0][l] = packw(Wsa2r[0][l], co2a[l], ci2a[l]);
                                  Wsa2[1][l] = packw(Wsa2r[1][l], co2b[l], ci2b[l]); }
    const int co3[3] = {128, 196, 256}, ci3[3] = {259, 128, 196};
    for (int l = 0; l < 3; ++l) { Wsa3[0][l] = packw(Wsa3r[0][l], co3[l], ci3[l]);
                                  Wsa3[1][l] = packw(Wsa3r[1][l], co3[l], ci3[l]); }
  }
  WP bin  = packw(bin_w, 512, 512),  bout = packw(bout_w, 512, 512);
  WP Wqkv[4], Wo[4], W1[4], W2[4];
  for (int b = 0; b < 4; ++b) {
    Wqkv[b] = packw(blk[b].Wqkv, 1536, 512);
    Wo[b]   = packw(blk[b].Wo, 512, 512);
    W1[b]   = packw(blk[b].W1, 2048, 512);
    W2[b]   = packw(blk[b].W2, 512, 2048);
  }
  WP fp3w[2] = { packw(fp3r[0], 256, 768), packw(fp3r[1], 256, 256) };
  WP fp2w[2] = { packw(fp2r[0], 256, 384), packw(fp2r[1], 128, 256) };
  WP fp1w[3] = { packw(fp1r[0], 128, 131), packw(fp1r[1], 128, 128), packw(fp1r[2], 128, 128) };
  WP h0 = packw(h0w, 128, 128), h1 = packw(h1w, 64, 128), ho = packw(how, 4, 64);

  tmp_base = off;

  // One SA branch: group -> 3x (WMMA gemm, BN scale, gelu) -> maxpool
  auto sa_branch = [&](const float* xyz_, int N_, const float* feat_, int Cf,
                       const float* nxyz, int ncent, float rad, int kk,
                       WP Wa, int ca, WP Wb, int cb, WP Wc, int cc,
                       float* outf, int ldo, int coff) {
    int Cin = 3 + Cf, Cp = an_al32(Cin);
    int cap = an_al32(ca), cbp = an_al32(cb), ccp = an_al32(cc);
    long P = (long)ncent * kk;
    float* grouped = alloc((size_t)P * Cp);
    float* g1 = alloc((size_t)P * cap);
    float* g2 = alloc((size_t)P * cbp);
    float* g3 = alloc((size_t)P * ccp);
    zero(grouped, (size_t)P * Cp);                    // K-pad cols must be 0
    if (cb & 31) zero(g2, (size_t)P * cbp);
    an_ball_group<<<(ncent + 255) / 256, 256, 0, stream>>>(xyz_, feat_, Cf, nxyz, ncent, N_,
                                                           rad * rad, kk, grouped, Cp);
    an_launch_gemm(stream, grouped, Cp, Wa.p, Wa.ld, g1, cap, (int)P, ca, Cin, nullptr, BN_S, 4);
    an_launch_gemm(stream, g1, cap, Wb.p, Wb.ld, g2, cbp, (int)P, cb, ca, nullptr, BN_S, 4);
    an_launch_gemm(stream, g2, cbp, Wc.p, Wc.ld, g3, ccp, (int)P, cc, cb, nullptr, BN_S, 4);
    an_maxpool<<<an_nblk((long)ncent * cc), 256, 0, stream>>>(g3, ccp, cc, ncent, kk, outf, ldo, coff);
  };

  auto run_sa = [&](const float* xyz_, int N_, const float* feat_, int Cf, int ncent,
                    float r0, int k0, float r1, int k1,
                    WP (*W)[3], const int (*co)[3],
                    int* idx_ws, float* nxyz, float* outf) {
    tmp_reset();
    an_fps<<<1, 256, 0, stream>>>(xyz_, N_, ncent, idx_ws, dist);
    an_gather_xyz<<<(ncent + 255) / 256, 256, 0, stream>>>(xyz_, idx_ws, nxyz, ncent);
    int ldo = co[0][2] + co[1][2];
    sa_branch(xyz_, N_, feat_, Cf, nxyz, ncent, r0, k0,
              W[0][0], co[0][0], W[0][1], co[0][1], W[0][2], co[0][2], outf, ldo, 0);
    sa_branch(xyz_, N_, feat_, Cf, nxyz, ncent, r1, k1,
              W[1][0], co[1][0], W[1][1], co[1][1], W[1][2], co[1][2], outf, ldo, co[0][2]);
  };

  // ---- Set abstraction ----
  const int co1[2][3] = { {32, 32, 64},  {32, 32, 64}  };
  const int co2[2][3] = { {64, 64, 128}, {64, 96, 128} };
  const int co3[2][3] = { {128, 196, 256}, {128, 196, 256} };
  run_sa(xyz, 16384, feats, 3, 4096, 0.05f, 16, 0.1f, 32, Wsa1, co1, idx1, l1x, l1f);
  run_sa(l1x, 4096, l1f, 128, 1024, 0.1f, 16, 0.2f, 32, Wsa2, co2, idx2, l2x, l2f);
  run_sa(l2x, 1024, l2f, 256, 256,  0.2f, 16, 0.4f, 32, Wsa3, co3, idx3p, l3x, l3f);

  // ---- Bottleneck in ----
  an_launch_gemm(stream, l3f, 512, bin.p, bin.ld, tok, 512, 256, 512, 512, bin_b, 1.0f, 0);

  // ---- Transformer blocks (tokens = 1 cond + 256; pads: rows->288, K 257->288) ----
  const int T = 257, TK = 288, TR = 288;
  float* x = tok;
  for (int b = 0; b < 4; ++b) {
    tmp_reset();
    float* xg  = (b & 1) ? xg1 : xg0;
    float* h   = alloc((size_t)TR * 512);
    float* qkv = alloc((size_t)TR * 1536);
    float* sc  = alloc((size_t)8 * TR * TK);
    float* vt  = alloc((size_t)512 * TK);
    float* ob  = alloc((size_t)TR * 512);
    float* f1  = alloc((size_t)TR * 2048);
    zero(sc, (size_t)8 * TR * TK);                      // K-pad cols of att must be 0
    an_gvec<<<2, 256, 0, stream>>>(gcond, blk[b].Wg, blk[b].bg, xg, 512);
    an_copy<<<512, 256, 0, stream>>>(x, xg + 512, 256L * 512);
    an_layernorm<<<T, 256, 0, stream>>>(xg, h, 512, blk[b].g1, blk[b].b1);
    an_launch_gemm(stream, h, 512, Wqkv[b].p, Wqkv[b].ld, qkv, 1536, T, 1536, 512, blk[b].bqkv, 1.0f, 0);
    for (int hh = 0; hh < 8; ++hh)  // scores = Q K^T / 8
      an_launch_gemm(stream, qkv + hh * 64, 1536, qkv + 512 + hh * 64, 1536,
                     sc + (size_t)hh * TR * TK, TK, T, T, 64, nullptr, 0.125f, 0);
    an_softmax<<<8 * T, 256, 0, stream>>>(sc, T, TK, (long)TR * TK, T);
    an_vt<<<an_nblk(512L * TK), 256, 0, stream>>>(qkv, 1536, T, TK, vt);
    for (int hh = 0; hh < 8; ++hh)  // o = att V  (via NT on transposed V)
      an_launch_gemm(stream, sc + (size_t)hh * TR * TK, TK, vt + (size_t)hh * 64 * TK, TK,
                     ob + hh * 64, 512, T, 64, T, nullptr, 1.0f, 0);
    an_launch_gemm(stream, ob, 512, Wo[b].p, Wo[b].ld, xg, 512, T, 512, 512, blk[b].bo, 1.0f, 2);
    an_layernorm<<<T, 256, 0, stream>>>(xg, h, 512, blk[b].g2, blk[b].b2);
    an_launch_gemm(stream, h, 512, W1[b].p, W1[b].ld, f1, 2048, T, 2048, 512, blk[b].bf1, 1.0f, 4);
    an_launch_gemm(stream, f1, 2048, W2[b].p, W2[b].ld, xg, 512, T, 512, 2048, blk[b].bf2, 1.0f, 2);
    x = xg + 512;  // drop cond row
  }

  // ---- Bottleneck out ----
  an_launch_gemm(stream, x, 512, bout.p, bout.ld, l3fn, 512, 256, 512, 512, bout_b, 1.0f, 0);

  // ---- Feature propagation ----
  auto run_fp = [&](const float* fx, int NF, const float* cx2, int NC,
                    const float* ff, int Cff, const float* fc, int Cfc,
                    const WP* Ws, const int* cos, int nlayers, float* outp) {
    tmp_reset();
    int* id3 = alloci((size_t)NF * 3);
    float* w3 = alloc((size_t)NF * 3);
    float* itp = alloc((size_t)NF * Cfc);
    an_3nn<<<(NF + 255) / 256, 256, 0, stream>>>(fx, NF, cx2, NC, id3, w3);
    an_interp<<<an_nblk((long)NF * Cfc), 256, 0, stream>>>(fc, Cfc, id3, w3, itp, (long)NF * Cfc);
    int Cin = Cff + Cfc, Cp = an_al32(Cin);
    float* cat = alloc((size_t)NF * Cp);
    if (Cin & 31) zero(cat, (size_t)NF * Cp);
    an_concat2<<<an_nblk((long)NF * Cin), 256, 0, stream>>>(ff, Cff, itp, Cfc, cat, Cp, NF);
    const float* cur = cat; int cld = Cp; int cc = Cin;
    for (int l = 0; l < nlayers; ++l) {
      float* nxt = (l == nlayers - 1) ? outp : alloc((size_t)NF * cos[l]);
      an_launch_gemm(stream, cur, cld, Ws[l].p, Ws[l].ld, nxt, cos[l], NF, cos[l], cc, nullptr, BN_S, 4);
      cur = nxt; cld = cos[l]; cc = cos[l];
    }
  };
  { const int c3[2] = {256, 256}; run_fp(l2x, 1024, l3x, 256, l2f, 256, l3fn, 512, fp3w, c3, 2, l2fn); }
  { const int c2[2] = {256, 128}; run_fp(l1x, 4096, l2x, 1024, l1f, 128, l2fn, 256, fp2w, c2, 2, l1fn); }
  { const int c1[3] = {128, 128, 128}; run_fp(xyz, 16384, l1x, 4096, feats, 3, l1fn, 128, fp1w, c1, 3, l0f); }

  // ---- Head: conv -> groupnorm(8) -> gelu, twice, then 64->4 ----
  an_launch_gemm(stream, l0f, 128, h0.p, h0.ld, t1, 128, 16384, 128, 128, h0b, 1.0f, 0);
  an_gn_stats<<<8, 256, 0, stream>>>(t1, 16384, 128, 8, gnst);
  an_gn_apply<<<an_nblk(16384L * 128), 256, 0, stream>>>(t1, t1, 16384, 128, 8, gnst, h0g, h0gb);
  an_launch_gemm(stream, t1, 128, h1.p, h1.ld, t2, 64, 16384, 64, 128, h1b, 1.0f, 0);
  an_gn_stats<<<8, 256, 0, stream>>>(t2, 16384, 64, 8, gnst);
  an_gn_apply<<<an_nblk(16384L * 64), 256, 0, stream>>>(t2, t2, 16384, 64, 8, gnst, h1g, h1gb);
  an_launch_gemm(stream, t2, 64, ho.p, ho.ld, (float*)d_out, 4, 16384, 4, 64, hob, 1.0f, 0);
}
